// Decoder_38981123179021
// MI455X (gfx1250) — matrix operand order repaired and checked by reading, not compiled, not run on hardware
//
#include <hip/hip_runtime.h>
#include <hip/hip_bf16.h>

#define DI __device__ __forceinline__

typedef __attribute__((ext_vector_type(16))) __bf16       v16bf;
typedef __attribute__((ext_vector_type(8)))  float        v8f;
typedef __attribute__((ext_vector_type(8)))  unsigned int v8u;

// ---------------- sizes ----------------
// B=32, Tenc=Tdec=600, TARGET=80, PRE=256, HID=512, ATT=128, LOCF=32, LOCK=31, POSTH=512
// ---------------- workspace layout (bytes, constexpr) ----------------
static constexpr size_t SZ_ENCBF = (size_t)32*600*512*2;
static constexpr size_t SZ_PM    = (size_t)19200*128*4;
static constexpr size_t SZ_DECP  = (size_t)19200*96*2;
static constexpr size_t SZ_H1    = (size_t)19200*256*2;
static constexpr size_t SZ_PRE   = (size_t)19200*256*2;
static constexpr size_t SZ_LOC   = (size_t)19200*128*4;

static constexpr size_t OFF_ENCBF=0;
static constexpr size_t OFF_PM   = OFF_ENCBF+SZ_ENCBF;
static constexpr size_t OFF_DECP = OFF_PM+SZ_PM;
static constexpr size_t OFF_H1   = OFF_DECP+SZ_DECP;
static constexpr size_t OFF_PRE  = OFF_H1+SZ_H1;
static constexpr size_t OFF_LOC  = OFF_PRE+SZ_PRE;
static constexpr size_t OFF_WPN1 = OFF_LOC+SZ_LOC;               // 256x96 bf16 (K padded 80->96)
static constexpr size_t OFF_WPN2 = OFF_WPN1+(size_t)256*96*2;    // 256x256 bf16
static constexpr size_t OFF_WM   = OFF_WPN2+(size_t)256*256*2;   // 128x512 bf16
static constexpr size_t OFF_WQ   = OFF_WM+(size_t)128*512*2;     // 128x512 bf16
static constexpr size_t OFF_WA   = OFF_WQ+(size_t)128*512*2;     // 2048x1280 bf16 [Wih|Whh]
static constexpr size_t OFF_BA   = OFF_WA+(size_t)2048*1280*2;   // 2048 f32 bih+bhh
static constexpr size_t OFF_WD   = OFF_BA+(size_t)2048*4;        // 2048x1536 bf16
static constexpr size_t OFF_BD   = OFF_WD+(size_t)2048*1536*2;   // 2048 f32
static constexpr size_t OFF_WLOC = OFF_BD+(size_t)2048*4;        // 128x64 bf16 merged conv*fc
static constexpr size_t OFF_WR   = OFF_WLOC+(size_t)128*64*2;    // 80x1024 bf16 proj reordered [ctx|dh]
static constexpr size_t OFF_WP0  = OFF_WR+(size_t)80*1024*2;     // 512x640 bf16 (5->8 taps)
static constexpr size_t OFF_WP1  = OFF_WP0+(size_t)512*640*2;    // 512x4096
static constexpr size_t OFF_WP2  = OFF_WP1+(size_t)512*4096*2;
static constexpr size_t OFF_WP3  = OFF_WP2+(size_t)512*4096*2;
static constexpr size_t OFF_WP4  = OFF_WP3+(size_t)512*4096*2;   // 80x4096
static constexpr size_t OFF_AH   = OFF_WP4+(size_t)80*4096*2;    // f32 states
static constexpr size_t OFF_AC   = OFF_AH+ (size_t)32*512*4;
static constexpr size_t OFF_DH   = OFF_AC+ (size_t)32*512*4;
static constexpr size_t OFF_DC   = OFF_DH+ (size_t)32*512*4;
static constexpr size_t OFF_CTX  = OFF_DC+ (size_t)32*512*4;
static constexpr size_t OFF_WCUM = OFF_CTX+(size_t)32*512*4;     // 32x600 f32
static constexpr size_t OFF_WPAD = OFF_WCUM+(size_t)32*600*4;    // 32x2x632 bf16 (padded [w;wcum])
static constexpr size_t OFF_XA   = OFF_WPAD+(size_t)32*2*632*2;  // 32x1280 bf16 [pre|ctx|ah]
static constexpr size_t OFF_XD   = OFF_XA+(size_t)32*1280*2;     // 32x1536 bf16 [ah|ctx|dh]
static constexpr size_t OFF_GATE = OFF_XD+(size_t)32*1536*2;     // 32x2048 f32
static constexpr size_t OFF_Q    = OFF_GATE+(size_t)32*2048*4;   // 32x128 f32
static constexpr size_t OFF_E    = OFF_Q+(size_t)32*128*4;       // 32x600 f32
static constexpr size_t OFF_MEL  = OFF_E+(size_t)32*600*4;       // 32x600x80 f32
static constexpr size_t WS_TOTAL = OFF_MEL+(size_t)32*600*80*4;  // ~95.9 MB

static constexpr size_t SZ_STATE = OFF_XA - OFF_AH;              // zero block AH..WPAD

// postnet ping-pong buffers aliased onto decoder-only scratch (dead after loop)
static constexpr size_t SZ_YP  = (size_t)32*512*608*2;           // (B, C, Tp=608) bf16
static constexpr size_t SZ_YP0 = (size_t)32*80*608*2;
static constexpr size_t OFF_YPA = 0;
static constexpr size_t OFF_YPB = SZ_YP;
static constexpr size_t OFF_YP0 = OFF_PRE;
static_assert(OFF_YPB + SZ_YP <= OFF_PRE, "ypB overlaps live postnet data");
static_assert(SZ_YP0 <= SZ_PRE, "yp0 overlaps live postnet data");

// ---------------- helpers ----------------
DI unsigned short f2b(float x){ unsigned u=__float_as_uint(x); return (unsigned short)((u+0x7fffu+((u>>16)&1u))>>16); }
DI float b2f(unsigned short b){ return __uint_as_float(((unsigned)b)<<16); }
DI float sigf(float x){ return 1.f/(1.f+__expf(-x)); }

DI v8f wmma_bf(v16bf a, v16bf b, v8f c){
  return __builtin_amdgcn_wmma_f32_16x16x32_bf16(false, a, false, b, (short)0, c, false, false);
}
#define V8F0 (v8f){0.f,0.f,0.f,0.f,0.f,0.f,0.f,0.f}

// A fragment (16x32 bf16, row-major, even lda): lanes0-15 K in {8h+0..7}, {16+8h+0..7}
DI v16bf frag_a(const unsigned short* A, int row0, int lda, int k0){
  int l=threadIdx.x&31, m=l&15, h=l>>4;
  const unsigned short* p = A + (size_t)(row0+m)*lda + k0;
  v8u r;
  #pragma unroll
  for(int v=0;v<4;++v) r[v]   = *(const unsigned*)(p + 8*h + 2*v);
  #pragma unroll
  for(int v=0;v<4;++v) r[4+v] = *(const unsigned*)(p + 16 + 8*h + 2*v);
  return __builtin_bit_cast(v16bf, r);
}
// B fragment: D=A@W^T, W row-major NxK: lane=column n, K in {8h+0..7}, {16+8h+0..7}
DI v16bf frag_b(const unsigned short* W, int n0, int ldw, int k0){
  int l=threadIdx.x&31, n=l&15, h=l>>4;
  const unsigned short* p = W + (size_t)(n0+n)*ldw + k0;
  v8u r;
  #pragma unroll
  for(int v=0;v<4;++v) r[v]   = *(const unsigned*)(p + 8*h + 2*v);
  #pragma unroll
  for(int v=0;v<4;++v) r[4+v] = *(const unsigned*)(p + 16 + 8*h + 2*v);
  return __builtin_bit_cast(v16bf, r);
}
// im2col A fragment for location conv: wpad (B,2,632), K=2ch x 32taps (tap 31 zero-weighted)
DI v16bf frag_a_loc(const unsigned short* wpad, int row0, int k0){
  int l=threadIdx.x&31, m=l&15, h=l>>4;
  int row=row0+m, b=row/600, t=row-600*b;
  v8u r;
  #pragma unroll
  for(int v=0;v<8;++v){
    int kp=(v<4)?(8*h+2*v):(16+8*h+2*(v-4));
    int k=k0+kp, ic=k>>5, kk=k&31;
    const unsigned short* p = wpad + (size_t)(b*2+ic)*632 + t + kk;   // may be 2B aligned
    r[v] = (unsigned)p[0] | ((unsigned)p[1]<<16);
  }
  return __builtin_bit_cast(v16bf, r);
}
// im2col A fragment for postnet conv: Y (B,Cin,608), K=Cin x 8taps (taps 5..7 zero-weighted)
DI v16bf frag_a_conv(const unsigned short* Y, int row0, int Cin, int k0){
  int l=threadIdx.x&31, m=l&15, h=l>>4;
  int row=row0+m, b=row/600, t=row-600*b;
  v8u r;
  #pragma unroll
  for(int v=0;v<8;++v){
    int kp=(v<4)?(8*h+2*v):(16+8*h+2*(v-4));
    int k=k0+kp, ci=k>>3, kk=k&7;
    const unsigned short* p = Y + (size_t)(b*Cin+ci)*608 + t + kk;
    r[v] = (unsigned)p[0] | ((unsigned)p[1]<<16);
  }
  return __builtin_bit_cast(v16bf, r);
}

// ---------------- setup kernels (grid-stride) ----------------
__global__ void k_f32_to_bf16(const float* s, unsigned short* d, int n){
  for(int i=blockIdx.x*blockDim.x+threadIdx.x;i<n;i+=gridDim.x*blockDim.x) d[i]=f2b(s[i]);
}
__global__ void k_add(const float* a, const float* b, float* o, int n){
  for(int i=blockIdx.x*blockDim.x+threadIdx.x;i<n;i+=gridDim.x*blockDim.x) o[i]=a[i]+b[i];
}
__global__ void k_build_decpad(const float* dec, unsigned short* o){ // rows (s*32+b), 96 cols, go-frame shift
  for(int i=blockIdx.x*blockDim.x+threadIdx.x;i<19200*96;i+=gridDim.x*blockDim.x){
    int row=i/96, k=i-96*row, s=row/32, b=row-32*s;
    float v=(s==0||k>=80)?0.f:dec[((size_t)b*600+(s-1))*80+k];
    o[i]=f2b(v);
  }
}
__global__ void k_build_wpn1(const float* w, unsigned short* o){ // 256x96, K pad
  for(int i=blockIdx.x*blockDim.x+threadIdx.x;i<256*96;i+=gridDim.x*blockDim.x){
    int n=i/96,k=i-96*n; o[i]=f2b(k<80?w[n*80+k]:0.f);
  }
}
__global__ void k_build_wcomb(const float* wih, const float* whh, unsigned short* o, int N,int Kih,int Khh){
  int K=Kih+Khh;
  for(int i=blockIdx.x*blockDim.x+threadIdx.x;i<N*K;i+=gridDim.x*blockDim.x){
    int n=i/K,k=i-K*n; o[i]=f2b(k<Kih?wih[(size_t)n*Kih+k]:whh[(size_t)n*Khh+(k-Kih)]);
  }
}
__global__ void k_build_wloc(const float* fc, const float* cw, unsigned short* o){ // merge conv(32,2,31)*fc(128,32)
  for(int i=blockIdx.x*blockDim.x+threadIdx.x;i<128*64;i+=gridDim.x*blockDim.x){
    int a=i>>6,k=i&63,ic=k>>5,kk=k&31; float s=0.f;
    if(kk<31) for(int f=0;f<32;++f) s+=fc[a*32+f]*cw[(f*2+ic)*31+kk];
    o[i]=f2b(s);
  }
}
__global__ void k_build_wr(const float* proj, unsigned short* o){ // reorder [dh|ctx] -> [ctx|dh]
  for(int i=blockIdx.x*blockDim.x+threadIdx.x;i<80*1024;i+=gridDim.x*blockDim.x){
    int n=i>>10,k=i&1023; o[i]=f2b(k<512?proj[n*1024+512+k]:proj[n*1024+(k-512)]);
  }
}
__global__ void k_build_wpost(const float* w, unsigned short* o, int Cout,int Cin){ // (Cout,Cin,5)->(Cout,Cin*8)
  int n=Cout*Cin*8;
  for(int i=blockIdx.x*blockDim.x+threadIdx.x;i<n;i+=gridDim.x*blockDim.x){
    int co=i/(Cin*8),r=i-co*Cin*8,ci=r>>3,kk=r&7;
    o[i]=f2b(kk<5?w[((size_t)co*Cin+ci)*5+kk]:0.f);
  }
}
__global__ void k_build_yp0(const float* mel, unsigned short* o){ // mel (B,T,80) -> ypad0 (B,80,608)
  for(int i=blockIdx.x*blockDim.x+threadIdx.x;i<32*80*600;i+=gridDim.x*blockDim.x){
    int b=i/(80*600),r=i-b*80*600,ci=r/600,t=r-600*ci;
    o[(size_t)(b*80+ci)*608+2+t]=f2b(mel[((size_t)b*600+t)*80+ci]);
  }
}

// ---------------- generic parallel WMMA GEMM (plain A) ----------------
// D = A(MxK bf16) @ W(NxK bf16)^T + bias ; out bf16(relu) or f32
__global__ void __launch_bounds__(256) gemm_plain(const unsigned short* A,int lda,
    const unsigned short* W,const float* bias,int M,int N,int K,
    float* outF,unsigned short* outB,int ldo,int relu){
  int ntiles=N>>4; int tile=blockIdx.x*8+(threadIdx.x>>5);
  if(tile>=(M>>4)*ntiles) return;
  int mt=tile/ntiles, nt=tile-mt*ntiles;
  v8f acc=V8F0;
  for(int ks=0;ks<K;ks+=32) acc=wmma_bf(frag_a(A,mt*16,lda,ks),frag_b(W,nt*16,K,ks),acc);
  int l=threadIdx.x&31,n=l&15,h=l>>4,col=nt*16+n;
  float bb=bias?bias[col]:0.f;
  #pragma unroll
  for(int v=0;v<8;++v){
    int row=mt*16+v+8*h; float r=acc[v]+bb;
    if(outB){ if(relu) r=fmaxf(r,0.f); outB[(size_t)row*ldo+col]=f2b(r); }
    else outF[(size_t)row*ldo+col]=r;
  }
}

// ---------------- postnet conv-as-GEMM (im2col A) ----------------
// outY!=0: ypad[b][co][2+t] = tanh(acc+bias)   (bf16)
// outY==0: d_out[row*80+col] = resid + acc + bias  (final layer, no tanh)
__global__ void __launch_bounds__(256) gemm_conv(const unsigned short* Y,int Cin,
    const unsigned short* W,const float* bias,int N,int K,
    unsigned short* outY,const float* resid,float* outF){
  int ntiles=N>>4; int tile=blockIdx.x*8+(threadIdx.x>>5);
  if(tile>=1200*ntiles) return;
  int mt=tile/ntiles, nt=tile-mt*ntiles;
  v8f acc=V8F0;
  for(int ks=0;ks<K;ks+=32) acc=wmma_bf(frag_a_conv(Y,mt*16,Cin,ks),frag_b(W,nt*16,K,ks),acc);
  int l=threadIdx.x&31,n=l&15,h=l>>4,col=nt*16+n;
  float bb=bias[col];
  #pragma unroll
  for(int v=0;v<8;++v){
    int row=mt*16+v+8*h, b=row/600, t=row-600*b;
    float r=acc[v]+bb;
    if(outY) outY[(size_t)(b*N+col)*608+2+t]=f2b(tanhf(r));
    else     outF[(size_t)row*80+col]=resid[(size_t)row*80+col]+r;
  }
}

// ---------------- persistent decoder recurrence: 1 WG, 32 wave32, 600 steps ----------------
DI void gsync(){ __threadfence(); __syncthreads(); __threadfence(); }

__global__ void __launch_bounds__(1024) k_decoder(char* ws, const float* vw){
  const unsigned short* pre  =(const unsigned short*)(ws+OFF_PRE);
  const unsigned short* encbf=(const unsigned short*)(ws+OFF_ENCBF);
  const float* pm  =(const float*)(ws+OFF_PM);
  const unsigned short* Wa=(const unsigned short*)(ws+OFF_WA);
  const unsigned short* Wd=(const unsigned short*)(ws+OFF_WD);
  const unsigned short* Wq=(const unsigned short*)(ws+OFF_WQ);
  const unsigned short* Wl=(const unsigned short*)(ws+OFF_WLOC);
  const unsigned short* Wr=(const unsigned short*)(ws+OFF_WR);
  const float* ba=(const float*)(ws+OFF_BA);
  const float* bd=(const float*)(ws+OFF_BD);
  float* ah=(float*)(ws+OFF_AH); float* ac=(float*)(ws+OFF_AC);
  float* dh=(float*)(ws+OFF_DH); float* dc=(float*)(ws+OFF_DC);
  float* ctx=(float*)(ws+OFF_CTX); float* wcum=(float*)(ws+OFF_WCUM);
  unsigned short* wpad=(unsigned short*)(ws+OFF_WPAD);
  unsigned short* xa=(unsigned short*)(ws+OFF_XA);
  unsigned short* xd=(unsigned short*)(ws+OFF_XD);
  float* gate=(float*)(ws+OFF_GATE); float* qb=(float*)(ws+OFF_Q);
  float* loc=(float*)(ws+OFF_LOC);  float* eb=(float*)(ws+OFF_E);
  float* mel=(float*)(ws+OFF_MEL);

  const int tid=threadIdx.x, wv=tid>>5, lane=tid&31;
  const int n_=lane&15, h_=lane>>4;

  for(int t=0;t<600;++t){
    // P0: xa = [pre_t | ctx | ah] bf16 ; refresh xd dh slice
    for(int i=tid;i<32*1280;i+=1024){
      int b=i/1280,k=i-1280*b; unsigned short v;
      if(k<256)      v=pre[((size_t)t*32+b)*256+k];
      else if(k<768) v=f2b(ctx[b*512+(k-256)]);
      else           v=f2b(ah[b*512+(k-768)]);
      xa[i]=v;
    }
    for(int i=tid;i<32*512;i+=1024){ int b=i>>9,j=i&511; xd[b*1536+1024+j]=f2b(dh[i]); }
    gsync();
    // P1: attention-LSTM gates = xa @ [Wih|Whh]^T + (bih+bhh)   M=32 N=2048 K=1280
    for(int idx=wv;idx<256;idx+=32){
      int mt=idx>>7,nt=idx&127; v8f acc=V8F0;
      for(int ks=0;ks<1280;ks+=32) acc=wmma_bf(frag_a(xa,mt*16,1280,ks),frag_b(Wa,nt*16,1280,ks),acc);
      int col=nt*16+n_; float bb=ba[col];
      #pragma unroll
      for(int v=0;v<8;++v) gate[(size_t)(mt*16+v+8*h_)*2048+col]=acc[v]+bb;
    }
    gsync();
    // P2: attention-LSTM pointwise
    for(int i=tid;i<32*512;i+=1024){
      int b=i>>9,j=i&511;
      float gi=gate[b*2048+j],gf=gate[b*2048+512+j],gg=gate[b*2048+1024+j],go=gate[b*2048+1536+j];
      float c2=sigf(gf)*ac[i]+sigf(gi)*tanhf(gg), hh=sigf(go)*tanhf(c2);
      ac[i]=c2; ah[i]=hh; xd[b*1536+j]=f2b(hh);
    }
    gsync();
    // P3: location GEMM (19200x128, K=64 merged conv*fc) + q GEMM (32x128, K=512)
    for(int idx=wv;idx<9616;idx+=32){
      if(idx<9600){
        int mt=idx>>3,nt=idx&7; v8f acc=V8F0;
        acc=wmma_bf(frag_a_loc(wpad,mt*16,0 ),frag_b(Wl,nt*16,64,0 ),acc);
        acc=wmma_bf(frag_a_loc(wpad,mt*16,32),frag_b(Wl,nt*16,64,32),acc);
        int col=nt*16+n_;
        #pragma unroll
        for(int v=0;v<8;++v) loc[(size_t)(mt*16+v+8*h_)*128+col]=acc[v];
      }else{
        int tq=idx-9600,mt=tq>>3,nt=tq&7; v8f acc=V8F0;
        for(int ks=0;ks<512;ks+=32) acc=wmma_bf(frag_a(xd,mt*16,1536,ks),frag_b(Wq,nt*16,512,ks),acc);
        int col=nt*16+n_;
        #pragma unroll
        for(int v=0;v<8;++v) qb[(mt*16+v+8*h_)*128+col]=acc[v];
      }
    }
    gsync();
    // P4: energies e = v . tanh(q + proc_mem + loc)
    for(int r=tid;r<19200;r+=1024){
      int b=r/600; const float* qp=qb+b*128; const float* pp=pm+(size_t)r*128; const float* lp=loc+(size_t)r*128;
      float s=0.f;
      for(int a2=0;a2<128;++a2) s+=vw[a2]*tanhf(qp[a2]+pp[a2]+lp[a2]);
      eb[r]=s;
    }
    gsync();
    // P5: wave b: softmax over 600, context, wcum, write padded bf16 [w;wcum]
    {
      int b=wv; const float* ep=eb+b*600;
      float mx=-3.4e38f;
      for(int tt=lane;tt<600;tt+=32) mx=fmaxf(mx,ep[tt]);
      #pragma unroll
      for(int o=16;o>0;o>>=1) mx=fmaxf(mx,__shfl_xor(mx,o,32));
      float s=0.f;
      for(int tt=lane;tt<600;tt+=32) s+=__expf(ep[tt]-mx);
      #pragma unroll
      for(int o=16;o>0;o>>=1) s+=__shfl_xor(s,o,32);
      float inv=1.f/s;
      float cacc[16];
      #pragma unroll
      for(int q2=0;q2<16;++q2) cacc[q2]=0.f;
      for(int t0=0;t0<600;t0+=32){
        int tt=t0+lane;
        float w2=(tt<600)?__expf(ep[tt]-mx)*inv:0.f;
        if(tt<600){
          float wc=wcum[b*600+tt]+w2; wcum[b*600+tt]=wc;
          wpad[(size_t)(b*2+0)*632+15+tt]=f2b(w2);
          wpad[(size_t)(b*2+1)*632+15+tt]=f2b(wc);
        }
        int lim=(600-t0<32)?(600-t0):32;
        for(int j=0;j<lim;++j){
          float wj=__shfl(w2,j,32);
          v8u u=*(const v8u*)(encbf+((size_t)b*600+t0+j)*512+lane*16);
          #pragma unroll
          for(int p2=0;p2<8;++p2){
            unsigned uu=u[p2];
            cacc[2*p2]  +=wj*__uint_as_float(uu<<16);
            cacc[2*p2+1]+=wj*__uint_as_float(uu&0xffff0000u);
          }
        }
      }
      #pragma unroll
      for(int q2=0;q2<16;++q2){
        float v=cacc[q2];
        ctx[b*512+lane*16+q2]=v;
        xd[b*1536+512+lane*16+q2]=f2b(v);
      }
    }
    gsync();
    // P6: decoder-LSTM gates = [ah|ctx|dh] @ [Wih|Whh]^T + bias   K=1536
    for(int idx=wv;idx<256;idx+=32){
      int mt=idx>>7,nt=idx&127; v8f acc=V8F0;
      for(int ks=0;ks<1536;ks+=32) acc=wmma_bf(frag_a(xd,mt*16,1536,ks),frag_b(Wd,nt*16,1536,ks),acc);
      int col=nt*16+n_; float bb=bd[col];
      #pragma unroll
      for(int v=0;v<8;++v) gate[(size_t)(mt*16+v+8*h_)*2048+col]=acc[v]+bb;
    }
    gsync();
    // P7: decoder-LSTM pointwise
    for(int i=tid;i<32*512;i+=1024){
      int b=i>>9,j=i&511;
      float gi=gate[b*2048+j],gf=gate[b*2048+512+j],gg=gate[b*2048+1024+j],go=gate[b*2048+1536+j];
      float c2=sigf(gf)*dc[i]+sigf(gi)*tanhf(gg), hh=sigf(go)*tanhf(c2);
      dc[i]=c2; dh[i]=hh; xd[b*1536+1024+j]=f2b(hh);
    }
    gsync();
    // P8: mel = [ctx|dh] @ Wr^T   M=32 N=80 K=1024  (A = xd+512, lda 1536)
    for(int idx=wv;idx<10;idx+=32){
      int mt=idx/5,nt=idx-5*mt; v8f acc=V8F0;
      for(int ks=0;ks<1024;ks+=32) acc=wmma_bf(frag_a(xd+512,mt*16,1536,ks),frag_b(Wr,nt*16,1024,ks),acc);
      int col=nt*16+n_;
      #pragma unroll
      for(int v=0;v<8;++v){ int b=mt*16+v+8*h_; mel[((size_t)b*600+t)*80+col]=acc[v]; }
    }
    gsync();
  }
}

// ---------------- host ----------------
extern "C" void kernel_launch(void* const* d_in, const int* in_sizes, int n_in,
                              void* d_out, int out_size, void* d_ws, size_t ws_size,
                              hipStream_t stream){
  (void)in_sizes; (void)n_in; (void)out_size; (void)ws_size;
  const float* enc  =(const float*)d_in[0];
  const float* dec  =(const float*)d_in[1];
  const float* pnw1 =(const float*)d_in[2];
  const float* pnb1 =(const float*)d_in[3];
  const float* pnw2 =(const float*)d_in[4];
  const float* pnb2 =(const float*)d_in[5];
  const float* awih =(const float*)d_in[6];
  const float* awhh =(const float*)d_in[7];
  const float* abih =(const float*)d_in[8];
  const float* abhh =(const float*)d_in[9];
  const float* dwih =(const float*)d_in[10];
  const float* dwhh =(const float*)d_in[11];
  const float* dbih =(const float*)d_in[12];
  const float* dbhh =(const float*)d_in[13];
  const float* memw =(const float*)d_in[14];
  const float* memb =(const float*)d_in[15];
  const float* qw   =(const float*)d_in[16];
  const float* cw   =(const float*)d_in[17];
  const float* fc   =(const float*)d_in[18];
  const float* vw   =(const float*)d_in[19];
  const float* proj =(const float*)d_in[20];
  const float* pw0=(const float*)d_in[21]; const float* pb0=(const float*)d_in[22];
  const float* pw1=(const float*)d_in[23]; const float* pb1=(const float*)d_in[24];
  const float* pw2=(const float*)d_in[25]; const float* pb2=(const float*)d_in[26];
  const float* pw3=(const float*)d_in[27]; const float* pb3=(const float*)d_in[28];
  const float* pw4=(const float*)d_in[29]; const float* pb4=(const float*)d_in[30];
  char* ws=(char*)d_ws;
  float* out=(float*)d_out;
  auto Wp=[&](size_t o){ return (unsigned short*)(ws+o); };
  auto Fp=[&](size_t o){ return (float*)(ws+o); };

  // setup: bf16 conversions, weight fusion/padding, state zeroing
  k_f32_to_bf16<<<4096,256,0,stream>>>(enc, Wp(OFF_ENCBF), 32*600*512);
  k_build_decpad<<<2048,256,0,stream>>>(dec, Wp(OFF_DECP));
  k_build_wpn1<<<96,256,0,stream>>>(pnw1, Wp(OFF_WPN1));
  k_f32_to_bf16<<<256,256,0,stream>>>(pnw2, Wp(OFF_WPN2), 256*256);
  k_f32_to_bf16<<<256,256,0,stream>>>(memw, Wp(OFF_WM), 128*512);
  k_f32_to_bf16<<<256,256,0,stream>>>(qw,   Wp(OFF_WQ), 128*512);
  k_build_wcomb<<<4096,256,0,stream>>>(awih, awhh, Wp(OFF_WA), 2048, 768, 512);
  k_build_wcomb<<<4096,256,0,stream>>>(dwih, dwhh, Wp(OFF_WD), 2048, 1024, 512);
  k_add<<<8,256,0,stream>>>(abih, abhh, Fp(OFF_BA), 2048);
  k_add<<<8,256,0,stream>>>(dbih, dbhh, Fp(OFF_BD), 2048);
  k_build_wloc<<<32,256,0,stream>>>(fc, cw, Wp(OFF_WLOC));
  k_build_wr<<<320,256,0,stream>>>(proj, Wp(OFF_WR));
  k_build_wpost<<<1024,256,0,stream>>>(pw0, Wp(OFF_WP0), 512, 80);
  k_build_wpost<<<4096,256,0,stream>>>(pw1, Wp(OFF_WP1), 512, 512);
  k_build_wpost<<<4096,256,0,stream>>>(pw2, Wp(OFF_WP2), 512, 512);
  k_build_wpost<<<4096,256,0,stream>>>(pw3, Wp(OFF_WP3), 512, 512);
  k_build_wpost<<<1024,256,0,stream>>>(pw4, Wp(OFF_WP4), 80, 512);
  (void)hipMemsetAsync(ws+OFF_AH, 0, SZ_STATE, stream);   // ah/ac/dh/dc/ctx/wcum/wpad

  // parallel pre-stages: prenet (2 GEMMs) + proc_mem GEMM (all WMMA bf16)
  gemm_plain<<<2400,256,0,stream>>>(Wp(OFF_DECP),96,  Wp(OFF_WPN1), pnb1, 19200,256,96,  nullptr, Wp(OFF_H1), 256, 1);
  gemm_plain<<<2400,256,0,stream>>>(Wp(OFF_H1), 256,  Wp(OFF_WPN2), pnb2, 19200,256,256, nullptr, Wp(OFF_PRE),256, 1);
  gemm_plain<<<1200,256,0,stream>>>(Wp(OFF_ENCBF),512,Wp(OFF_WM),   memb, 19200,128,512, Fp(OFF_PM), nullptr,128, 0);

  // sequential 600-step recurrence: single persistent workgroup
  k_decoder<<<1,1024,0,stream>>>(ws, vw);

  // postnet: 5 conv layers as channel-padded im2col WMMA GEMMs (ping-pong aliased buffers)
  (void)hipMemsetAsync(ws+OFF_YP0, 0, SZ_YP0, stream);
  (void)hipMemsetAsync(ws+OFF_YPA, 0, SZ_YP,  stream);
  (void)hipMemsetAsync(ws+OFF_YPB, 0, SZ_YP,  stream);
  k_build_yp0<<<4096,256,0,stream>>>(Fp(OFF_MEL), Wp(OFF_YP0));
  gemm_conv<<<4800,256,0,stream>>>(Wp(OFF_YP0), 80,  Wp(OFF_WP0), pb0, 512, 640,  Wp(OFF_YPA), nullptr, nullptr);
  gemm_conv<<<4800,256,0,stream>>>(Wp(OFF_YPA), 512, Wp(OFF_WP1), pb1, 512, 4096, Wp(OFF_YPB), nullptr, nullptr);
  gemm_conv<<<4800,256,0,stream>>>(Wp(OFF_YPB), 512, Wp(OFF_WP2), pb2, 512, 4096, Wp(OFF_YPA), nullptr, nullptr);
  gemm_conv<<<4800,256,0,stream>>>(Wp(OFF_YPA), 512, Wp(OFF_WP3), pb3, 512, 4096, Wp(OFF_YPB), nullptr, nullptr);
  gemm_conv<<<750, 256,0,stream>>>(Wp(OFF_YPB), 512, Wp(OFF_WP4), pb4, 80,  4096, nullptr, Fp(OFF_MEL), out);
}
